// MinModel_67388036874289
// MI455X (gfx1250) — matrix operand order repaired and checked by reading, not compiled, not run on hardware
//
#include <hip/hip_runtime.h>
#include <hip/hip_bf16.h>

#define T_DIM 1024
#define V_DIM 2048
#define B_DIM 8

#define A_STRIDE 40    // 64 rows x 32 cols bf16, padded (80B rows)
#define B_STRIDE 136   // 32 rows x 128 cols bf16, padded (272B rows)

typedef __attribute__((ext_vector_type(16))) __bf16 v16bf;
typedef __attribute__((ext_vector_type(8)))  float  v8f;

// exact parameter type of the async-LDS builtins (from clang diagnostic):
// 'int __attribute__((vector_size(16)))' pointers in AS1 (global) / AS3 (LDS)
typedef int b128_t __attribute__((vector_size(16)));

union FragAB {
    v16bf v;
    uint4 u[2];
};

// ---------------------------------------------------------------------------
// CDNA5 async global->LDS copy (ASYNCcnt-tracked). Guarded: falls back to
// synchronous register staging if the toolchain lacks the builtins.
// ---------------------------------------------------------------------------
#if defined(__has_builtin)
#if __has_builtin(__builtin_amdgcn_global_load_async_to_lds_b128)
#define HAVE_ASYNC_LDS 1
#endif
#endif

#ifdef HAVE_ASYNC_LDS
#define COPY_B128(gsrc, ldst)                                                  \
    __builtin_amdgcn_global_load_async_to_lds_b128(                            \
        (__attribute__((address_space(1))) b128_t*)(gsrc),                     \
        (__attribute__((address_space(3))) b128_t*)(ldst), 0, 0)
#if __has_builtin(__builtin_amdgcn_s_wait_asynccnt)
#define WAIT_ASYNC(n) __builtin_amdgcn_s_wait_asynccnt(n)
#else
#define WAIT_ASYNC(n) asm volatile("s_wait_asynccnt %0" ::"i"(n) : "memory")
#endif
#else
#define COPY_B128(gsrc, ldst)                                                  \
    do { *(uint4*)(ldst) = *(const uint4*)(gsrc); } while (0)
#define WAIT_ASYNC(n) do { } while (0)
#endif

// ---------------------------------------------------------------------------
// Block reduction helpers (blockDim.x == 256)
// ---------------------------------------------------------------------------
__device__ __forceinline__ float blockReduceMax(float v, float* smem) {
    int tid = threadIdx.x;
    smem[tid] = v;
    __syncthreads();
    for (int off = 128; off > 0; off >>= 1) {
        if (tid < off) smem[tid] = fmaxf(smem[tid], smem[tid + off]);
        __syncthreads();
    }
    float r = smem[0];
    __syncthreads();
    return r;
}

__device__ __forceinline__ float blockReduceSum(float v, float* smem) {
    int tid = threadIdx.x;
    smem[tid] = v;
    __syncthreads();
    for (int off = 128; off > 0; off >>= 1) {
        if (tid < off) smem[tid] += smem[tid + off];
        __syncthreads();
    }
    float r = smem[0];
    __syncthreads();
    return r;
}

// ---------------------------------------------------------------------------
// Kernel 1: a1T[s', t] = softmax_s( v_weight[t-s] | s<=t )[s'] ; 0 for s'>t.
// Stored transposed + bf16 so it can be used directly as the WMMA B operand.
// ---------------------------------------------------------------------------
__global__ __launch_bounds__(256) void a1T_kernel(const float* __restrict__ vw,
                                                  __hip_bfloat16* __restrict__ a1T) {
    __shared__ float smem[256];
    const int t   = blockIdx.x;
    const int tid = threadIdx.x;

    float m = -3.402823466e+38f;
    for (int r = tid; r <= t; r += 256) m = fmaxf(m, vw[r]);
    m = blockReduceMax(m, smem);

    float z = 0.f;
    for (int r = tid; r <= t; r += 256) z += __expf(vw[r] - m);
    z = blockReduceSum(z, smem);
    const float invz = 1.0f / z;

    for (int s = tid; s < T_DIM; s += 256) {
        float val = (s <= t) ? __expf(vw[t - s] - m) * invz : 0.0f;
        a1T[(size_t)s * T_DIM + t] = __float2bfloat16(val);
    }
}

// ---------------------------------------------------------------------------
// Kernel 2: Wg[b][t][s'] = (bf16) W[idx[b,t], idx[b,s']]   (gathered A matrix)
// ---------------------------------------------------------------------------
__global__ __launch_bounds__(256) void gather_kernel(const int* __restrict__ idx,
                                                     const float* __restrict__ W,
                                                     __hip_bfloat16* __restrict__ Wg) {
    const int bt = blockIdx.x;          // b*T + t
    const int b  = bt >> 10;
    const int* idxb = idx + (size_t)b * T_DIM;
    const int row = idx[bt];
    const float* wrow = W + (size_t)row * V_DIM;
    __hip_bfloat16* out = Wg + (size_t)bt * T_DIM;
    for (int s = threadIdx.x; s < T_DIM; s += 256) {
        out[s] = __float2bfloat16(wrow[idxb[s]]);
    }
}

// ---------------------------------------------------------------------------
// Kernel 3: a2[b] = Wg[b] @ a1T, bf16 WMMA with fp32 accumulation.
// Block = 8 waves (2x4), block tile 64x128, wave tile 32x32 (2x2 frags).
// Global->LDS staging uses GLOBAL_LOAD_ASYNC_TO_LDS_B128 (ASYNCcnt),
// double-buffered: issue stage k+1, s_wait_asynccnt<=3, barrier, consume k.
// Fragment addressing follows ISA 7.12.2 wave32 layouts.
// ---------------------------------------------------------------------------
__device__ __forceinline__ void stage_copy(const __hip_bfloat16* __restrict__ Ab,
                                           const __hip_bfloat16* __restrict__ Bt,
                                           __hip_bfloat16* As, __hip_bfloat16* Bs,
                                           int mBlk, int nBlk, int k0, int tid) {
    // A tile: 64 rows x 32 cols, 16B chunks -> 256 chunks, 1 per thread
    {
        const int row = tid >> 2, cc = tid & 3;
        const __hip_bfloat16* g = Ab + (size_t)(mBlk + row) * T_DIM + k0 + cc * 8;
        COPY_B128(g, As + row * A_STRIDE + cc * 8);
        __builtin_prefetch((const void*)(g + 64), 0, 1);  // L2 prefetch, two slabs ahead
    }
    // B tile: 32 rows x 128 cols, 16B chunks -> 512 chunks, 2 per thread
#pragma unroll
    for (int i = 0; i < 2; ++i) {
        const int c = tid + i * 256;
        const int row = c >> 4, cc = c & 15;
        const __hip_bfloat16* g = Bt + (size_t)(k0 + row) * T_DIM + nBlk + cc * 8;
        COPY_B128(g, Bs + row * B_STRIDE + cc * 8);
    }
}

__global__ __launch_bounds__(256) void gemm_a2_kernel(const __hip_bfloat16* __restrict__ Wg,
                                                      const __hip_bfloat16* __restrict__ a1T,
                                                      float* __restrict__ a2) {
    __shared__ __align__(16) __hip_bfloat16 Atile[2][64 * A_STRIDE];
    __shared__ __align__(16) __hip_bfloat16 Btile[2][32 * B_STRIDE];

    const int tid  = threadIdx.x;
    const int lane = tid & 31;
    const int wave = tid >> 5;
    const int h    = lane >> 4;
    const int l16  = lane & 15;
    const int wm   = wave >> 2;            // 0..1
    const int wn   = wave & 3;             // 0..3
    const int b    = blockIdx.z;
    const int mBlk = blockIdx.y * 64;
    const int nBlk = blockIdx.x * 128;

    const __hip_bfloat16* Ab = Wg + (size_t)b * T_DIM * T_DIM;
    float* Cb = a2 + (size_t)b * T_DIM * T_DIM;

    v8f acc[2][2];
#pragma unroll
    for (int mi = 0; mi < 2; ++mi)
#pragma unroll
        for (int ni = 0; ni < 2; ++ni)
#pragma unroll
            for (int e = 0; e < 8; ++e) acc[mi][ni][e] = 0.0f;

    int buf = 0;
    stage_copy(Ab, a1T, Atile[0], Btile[0], mBlk, nBlk, 0, tid);

    for (int k0 = 0; k0 < T_DIM; k0 += 32) {
        const int nxt = buf ^ 1;
        if (k0 + 32 < T_DIM) {
            stage_copy(Ab, a1T, Atile[nxt], Btile[nxt], mBlk, nBlk, k0 + 32, tid);
            WAIT_ASYNC(3);   // 3 async ops/wave for next stage still in flight
        } else {
            WAIT_ASYNC(0);
        }
        __syncthreads();     // all waves' copies for `buf` visible

        const __hip_bfloat16* As = Atile[buf];
        const __hip_bfloat16* Bs = Btile[buf];

        FragAB aF[2], bF[2];
#pragma unroll
        for (int mi = 0; mi < 2; ++mi) {
            const __hip_bfloat16* p = As + (wm * 32 + mi * 16 + l16) * A_STRIDE + 8 * h;
            aF[mi].u[0] = *(const uint4*)(p);        // K = 8h   .. 8h+7
            aF[mi].u[1] = *(const uint4*)(p + 16);   // K = 16+8h.. +7
        }
#pragma unroll
        for (int ni = 0; ni < 2; ++ni) {
            const unsigned short* p = (const unsigned short*)(Bs + (8 * h) * B_STRIDE + wn * 32 + ni * 16 + l16);
#pragma unroll
            for (int s = 0; s < 16; ++s)             // K = 8h..8h+7, 16+8h..+7
                bF[ni].v[s] = __builtin_bit_cast(__bf16, p[(s < 8 ? s : s + 8) * B_STRIDE]);
        }

#pragma unroll
        for (int mi = 0; mi < 2; ++mi)
#pragma unroll
            for (int ni = 0; ni < 2; ++ni)
                acc[mi][ni] = __builtin_amdgcn_wmma_f32_16x16x32_bf16(
                    false, aF[mi].v, false, bF[ni].v,
                    (short)0, acc[mi][ni], false, false);

        __syncthreads();     // reads of `buf` done before it is overwritten
        buf = nxt;
    }

#pragma unroll
    for (int mi = 0; mi < 2; ++mi)
#pragma unroll
        for (int ni = 0; ni < 2; ++ni)
#pragma unroll
            for (int r = 0; r < 8; ++r) {
                Cb[(size_t)(mBlk + wm * 32 + mi * 16 + r + 8 * h) * T_DIM +
                   (nBlk + wn * 32 + ni * 16 + l16)] = acc[mi][ni][r];
            }
}

// ---------------------------------------------------------------------------
// Kernel 4: causal softmax of a2 rows: row (b,t) over s in [0, t].
// ---------------------------------------------------------------------------
__global__ __launch_bounds__(256) void softmax_kernel(float* __restrict__ a2) {
    __shared__ float smem[256];
    const int bt  = blockIdx.x;
    const int t   = bt & (T_DIM - 1);
    const int tid = threadIdx.x;
    float* row = a2 + (size_t)bt * T_DIM;

    float m = -3.402823466e+38f;
    for (int s = tid; s <= t; s += 256) m = fmaxf(m, row[s]);
    m = blockReduceMax(m, smem);

    float z = 0.f;
    for (int s = tid; s <= t; s += 256) z += __expf(row[s] - m);
    z = blockReduceSum(z, smem);
    const float invz = 1.0f / z;

    for (int s = tid; s <= t; s += 256) row[s] = __expf(row[s] - m) * invz;
}

// ---------------------------------------------------------------------------
// Kernel 5: logits[b,t,v] = sum_{s<=t} a2[b,t,s] * [idx[b,s]==v]
// LDS accumulator (ds atomics), then coalesced row store.
// ---------------------------------------------------------------------------
__global__ __launch_bounds__(256) void scatter_kernel(const int* __restrict__ idx,
                                                      const float* __restrict__ a2,
                                                      float* __restrict__ out) {
    __shared__ float accs[V_DIM];
    const int bt  = blockIdx.x;
    const int b   = bt >> 10;
    const int t   = bt & (T_DIM - 1);
    const int tid = threadIdx.x;

    for (int v = tid; v < V_DIM; v += 256) accs[v] = 0.0f;
    __syncthreads();

    const int* idxb = idx + (size_t)b * T_DIM;
    const float* row = a2 + (size_t)bt * T_DIM;
    for (int s = tid; s <= t; s += 256) {
        atomicAdd(&accs[idxb[s]], row[s]);
    }
    __syncthreads();

    float* orow = out + (size_t)bt * V_DIM;
    for (int v = tid; v < V_DIM; v += 256) orow[v] = accs[v];
}

// ---------------------------------------------------------------------------
// Launch
// ---------------------------------------------------------------------------
extern "C" void kernel_launch(void* const* d_in, const int* in_sizes, int n_in,
                              void* d_out, int out_size, void* d_ws, size_t ws_size,
                              hipStream_t stream) {
    (void)in_sizes; (void)n_in; (void)out_size; (void)ws_size;

    const int*   idx = (const int*)d_in[0];    // [B,T]
    const float* vw  = (const float*)d_in[1];  // [T,1]
    const float* W   = (const float*)d_in[2];  // [V,V]
    float*       out = (float*)d_out;          // [B,T,V]

    char* ws = (char*)d_ws;
    __hip_bfloat16* a1T = (__hip_bfloat16*)ws;                                 // 2 MB
    __hip_bfloat16* Wg  = (__hip_bfloat16*)(ws + (size_t)T_DIM * T_DIM * 2);   // 16 MB
    float*          a2  = (float*)(ws + (size_t)T_DIM * T_DIM * 2
                                      + (size_t)B_DIM * T_DIM * T_DIM * 2);    // 33.5 MB

    a1T_kernel<<<T_DIM, 256, 0, stream>>>(vw, a1T);
    gather_kernel<<<B_DIM * T_DIM, 256, 0, stream>>>(idx, W, Wg);

    dim3 ggrid(T_DIM / 128, T_DIM / 64, B_DIM);
    gemm_a2_kernel<<<ggrid, 256, 0, stream>>>(Wg, a1T, a2);

    softmax_kernel<<<B_DIM * T_DIM, 256, 0, stream>>>(a2);
    scatter_kernel<<<B_DIM * T_DIM, 256, 0, stream>>>(idx, a2, out);
}
